// ISTFT_21380347199930
// MI455X (gfx1250) — matrix operand order repaired and checked by reading, not compiled, not run on hardware
//
#include <hip/hip_runtime.h>
#include <hip/hip_bf16.h>

typedef __attribute__((ext_vector_type(16))) _Float16 v16h;
typedef __attribute__((ext_vector_type(8)))  _Float16 v8h;
typedef __attribute__((ext_vector_type(8)))  float    v8f;
typedef int v4i_ __attribute__((vector_size(16)));     // matches builtin param type

#define N_FFT   2048
#define HOP     512
#define NFRAMES 2048
#define NBINS   1025
#define KPAD    2112              // 2*1056; 1056 = 1025 bins padded to mult of 32
#define KHALF   1056
#define OUT_LEN 1048064           // HOP*(NFRAMES-1)
#define NBATCH  8
#define TWO_PI  6.28318530717958647692f

#if defined(__AMDGCN__) && __has_builtin(__builtin_amdgcn_global_load_async_to_lds_b128)
#define HAVE_ASYNC_LDS 1
#endif

// ---------------------------------------------------------------------------
// Kernel 1: build fused [Ar | Ai] basis, f16, shape [2048][KPAD]
//   Ar[a,k] = c_k * cos(2*pi*a*k/N) * win[a]/N   (k<=1024; c_0=c_1024=1 else 2)
//   Ai[a,k] = -2 * sin(2*pi*a*k/N) * win[a]/N    (1<=k<=1023, else 0)
// Exact argument reduction via (a*k) mod 2048.
// ---------------------------------------------------------------------------
__global__ __launch_bounds__(256) void build_basis_kernel(_Float16* __restrict__ basis) {
    int idx = blockIdx.x * 256 + threadIdx.x;
    if (idx >= NFRAMES * KPAD) return;
    int a = idx / KPAD;
    int k = idx % KPAD;
    float win = 0.5f - 0.5f * cosf(TWO_PI * (float)a * (1.0f / 2048.0f));
    float c = 0.0f;
    if (k < KHALF) {
        int kk = k;
        if (kk <= 1024) {
            int p = (a * kk) & (N_FFT - 1);
            float ang = TWO_PI * (float)p * (1.0f / 2048.0f);
            float ck = (kk == 0 || kk == 1024) ? 1.0f : 2.0f;
            c = ck * cosf(ang);
        }
    } else {
        int kk = k - KHALF;
        if (kk >= 1 && kk <= 1023) {
            int p = (a * kk) & (N_FFT - 1);
            float ang = TWO_PI * (float)p * (1.0f / 2048.0f);
            c = -2.0f * sinf(ang);
        }
    }
    basis[idx] = (_Float16)(c * win * (1.0f / 2048.0f));
}

// ---------------------------------------------------------------------------
// Kernel 2: pack spectra into fused f16 K layout [b*NFRAMES + t][KPAD]:
//   k in [0,1056)      -> real bin k   (0 for k>1024)
//   k in [1056,2112)   -> imag bin k-1056 (0 for kk>1024)
// One bandwidth pass; removes all conditionals from the GEMM inner loop.
// ---------------------------------------------------------------------------
__global__ __launch_bounds__(256) void pack_spec_kernel(
    const float* __restrict__ real, const float* __restrict__ imag,
    _Float16* __restrict__ spec)
{
    long long idx = (long long)blockIdx.x * 256 + threadIdx.x;
    if (idx >= (long long)NBATCH * NFRAMES * KPAD) return;
    int k = (int)(idx % KPAD);
    long long bt = idx / KPAD;                 // b*NFRAMES + t
    float v;
    if (k < KHALF) {
        v = (k <= 1024) ? real[bt * NBINS + k] : 0.0f;
    } else {
        int kk = k - KHALF;
        v = (kk <= 1024) ? imag[bt * NBINS + kk] : 0.0f;
    }
    spec[idx] = (_Float16)v;
}

// ---------------------------------------------------------------------------
// 64-byte global->LDS tile copy: async data mover if available, else vector copy.
// ---------------------------------------------------------------------------
__device__ __forceinline__ void copy64_to_lds(const _Float16* g, _Float16* l) {
#if defined(HAVE_ASYNC_LDS)
    __attribute__((address_space(1))) v4i_* gp =
        (__attribute__((address_space(1))) v4i_*)g;
    __attribute__((address_space(3))) v4i_* lp =
        (__attribute__((address_space(3))) v4i_*)l;
    __builtin_amdgcn_global_load_async_to_lds_b128(gp, lp, 0, 0);
    __builtin_amdgcn_global_load_async_to_lds_b128(gp, lp, 16, 0);
    __builtin_amdgcn_global_load_async_to_lds_b128(gp, lp, 32, 0);
    __builtin_amdgcn_global_load_async_to_lds_b128(gp, lp, 48, 0);
#else
    const uint4* gs = (const uint4*)g;
    uint4* d = (uint4*)l;
    uint4 t0 = gs[0], t1 = gs[1], t2 = gs[2], t3 = gs[3];
    d[0] = t0; d[1] = t1; d[2] = t2; d[3] = t3;
#endif
}

__device__ __forceinline__ void wait_lds_copies() {
#if defined(HAVE_ASYNC_LDS)
#if __has_builtin(__builtin_amdgcn_s_wait_asynccnt)
    __builtin_amdgcn_s_wait_asynccnt(0);
#else
    asm volatile("s_wait_asynccnt 0" ::: "memory");
#endif
#endif
}

// ---------------------------------------------------------------------------
// Kernel 3: per-batch GEMM  S[b] (2048x2048) = Spec[b] (2048xKPAD) * Basis^T
//   Block: 128(M=frames) x 128(N=samples) tile, 256 threads = 8 wave32.
//   Wave (wm 0..3, wn 0..1) computes 32x64 = 2x4 WMMA 16x16 tiles.
//   K staged in 64-chunks through LDS (rows padded to 72 halves -> the
//   per-lane b128 fragment reads are bank-conflict-free).
// ---------------------------------------------------------------------------
#define LDS_STRIDE 72

__global__ __launch_bounds__(256) void istft_gemm_kernel(
    const _Float16* __restrict__ spec, const _Float16* __restrict__ basis,
    _Float16* __restrict__ S)
{
    __shared__ alignas(16) _Float16 smA[128 * LDS_STRIDE];
    __shared__ alignas(16) _Float16 smB[128 * LDS_STRIDE];

    const int tid  = threadIdx.x;
    const int lane = tid & 31;
    const int wave = tid >> 5;
    const int wm   = wave & 3;   // 4-way split of M (32 rows each)
    const int wn   = wave >> 2;  // 2-way split of N (64 cols each)
    const int b    = blockIdx.z;
    const int Mb   = blockIdx.y * 128;   // frame-tile base
    const int Nb   = blockIdx.x * 128;   // sample-tile base

    v8f acc[2][4];
#pragma unroll
    for (int mt = 0; mt < 2; ++mt)
#pragma unroll
        for (int nt = 0; nt < 4; ++nt)
            acc[mt][nt] = (v8f){0.f,0.f,0.f,0.f,0.f,0.f,0.f,0.f};

    // staging role: 2 threads per row, 32 halves (64 B) each
    const int row  = tid >> 1;            // 0..127
    const int kseg = (tid & 1) * 32;      // 0 or 32
    const _Float16* Arow = spec  + ((size_t)(b * NFRAMES + Mb + row)) * KPAD + kseg;
    const _Float16* Brow = basis + ((size_t)(Nb + row)) * KPAD + kseg;
    _Float16* ldsA = &smA[row * LDS_STRIDE + kseg];
    _Float16* ldsB = &smB[row * LDS_STRIDE + kseg];

    const int hl  = lane >> 4;            // half-wave (0/1)
    const int l15 = lane & 15;

    for (int kc = 0; kc < KPAD; kc += 64) {
        copy64_to_lds(Arow + kc, ldsA);
        copy64_to_lds(Brow + kc, ldsB);
        if (kc + 64 < KPAD) {
            __builtin_prefetch(Arow + kc + 64, 0, 0);   // global_prefetch_b8
            __builtin_prefetch(Brow + kc + 64, 0, 0);
        }
        wait_lds_copies();
        __syncthreads();

        // ---- two WMMA K-steps per stage
#pragma unroll
        for (int kk = 0; kk < 64; kk += 32) {
            v16h aF[2], bF[4];
#pragma unroll
            for (int mt = 0; mt < 2; ++mt) {
                // 16-bit A layout: lane holds row M=l15; K = hl*8+{0..7} and +16
                const _Float16* p =
                    &smA[(wm * 32 + mt * 16 + l15) * LDS_STRIDE + kk + (hl << 3)];
                v8h lo = *(const v8h*)p;
                v8h hi = *(const v8h*)(p + 16);
                aF[mt] = __builtin_shufflevector(lo, hi,
                         0,1,2,3,4,5,6,7,8,9,10,11,12,13,14,15);
            }
#pragma unroll
            for (int nt = 0; nt < 4; ++nt) {
                // B layout: lane holds column N=l15; K = hl*8+{0..7} and +16
                const _Float16* p =
                    &smB[(wn * 64 + nt * 16 + l15) * LDS_STRIDE + kk + (hl << 3)];
                v8h lo = *(const v8h*)p;
                v8h hi = *(const v8h*)(p + 16);
                bF[nt] = __builtin_shufflevector(lo, hi,
                         0,1,2,3,4,5,6,7,8,9,10,11,12,13,14,15);
            }
#pragma unroll
            for (int mt = 0; mt < 2; ++mt)
#pragma unroll
                for (int nt = 0; nt < 4; ++nt)
                    acc[mt][nt] = __builtin_amdgcn_wmma_f32_16x16x32_f16(
                        false, aF[mt], false, bF[nt], (short)0, acc[mt][nt],
                        false, false);
        }
        __syncthreads();
    }

    // ---- store D tiles to f16 staging: C/D layout VGPR j -> M = j + 8*hl, N = l15
#pragma unroll
    for (int mt = 0; mt < 2; ++mt) {
#pragma unroll
        for (int nt = 0; nt < 4; ++nt) {
            int m0 = Mb + wm * 32 + mt * 16 + (hl << 3);
            int n0 = Nb + wn * 64 + nt * 16 + l15;
#pragma unroll
            for (int j = 0; j < 8; ++j) {
                S[((size_t)(b * NFRAMES + m0 + j)) * N_FFT + n0] =
                    (_Float16)acc[mt][nt][j];
            }
        }
    }
}

// ---------------------------------------------------------------------------
// Kernel 4: deterministic overlap-add gather + window-sumsquare normalization.
// Output sample n (post-slice) = pre-slice index m = n + 1024; contributions
// from frames t with 0 <= m - 512 t < 2048 (<=4 of them).
// ---------------------------------------------------------------------------
__global__ __launch_bounds__(256) void istft_epilogue_kernel(
    const _Float16* __restrict__ S, float* __restrict__ out)
{
    long long idx = (long long)blockIdx.x * 256 + threadIdx.x;
    if (idx >= (long long)NBATCH * OUT_LEN) return;
    int b = (int)(idx / OUT_LEN);
    int n = (int)(idx % OUT_LEN);
    int m = n + N_FFT / 2;                       // center=True slice offset

    int t1 = m >> 9;                  if (t1 > NFRAMES - 1) t1 = NFRAMES - 1;
    int t0 = (m - (N_FFT - 1) + (HOP - 1)) >> 9; if (t0 < 0) t0 = 0;

    float s = 0.0f, w2 = 0.0f;
    for (int t = t0; t <= t1; ++t) {
        int a = m - (t << 9);                    // 0..2047
        s += (float)S[((size_t)(b * NFRAMES + t)) * N_FFT + a];
        float w = 0.5f - 0.5f * cosf(TWO_PI * (float)a * (1.0f / 2048.0f));
        w2 += w * w;
    }
    out[idx] = s / fmaxf(w2, 1e-8f);
}

// ---------------------------------------------------------------------------
extern "C" void kernel_launch(void* const* d_in, const int* in_sizes, int n_in,
                              void* d_out, int out_size, void* d_ws, size_t ws_size,
                              hipStream_t stream) {
    (void)in_sizes; (void)n_in; (void)out_size; (void)ws_size;
    const float* real = (const float*)d_in[0];
    const float* imag = (const float*)d_in[1];
    float* out = (float*)d_out;

    char* ws = (char*)d_ws;
    _Float16* basis  = (_Float16*)ws;                        // 2048*2112*2   =  8,650,752 B
    _Float16* spec   = (_Float16*)(ws + 8650752);            // 8*2048*2112*2 = 69,206,016 B
    _Float16* Sstage = (_Float16*)(ws + 8650752 + 69206016); // 8*2048*2048*2 = 67,108,864 B

    {
        int total = NFRAMES * KPAD;
        build_basis_kernel<<<(total + 255) / 256, 256, 0, stream>>>(basis);
    }
    {
        long long total = (long long)NBATCH * NFRAMES * KPAD;   // 34,603,008
        pack_spec_kernel<<<(unsigned)((total + 255) / 256), 256, 0, stream>>>(
            real, imag, spec);
    }
    {
        dim3 grid(N_FFT / 128, NFRAMES / 128, NBATCH);          // (16,16,8)
        istft_gemm_kernel<<<grid, 256, 0, stream>>>(spec, basis, Sstage);
    }
    {
        long long total = (long long)NBATCH * OUT_LEN;          // 8,384,512
        istft_epilogue_kernel<<<(unsigned)((total + 255) / 256), 256, 0, stream>>>(
            Sstage, out);
    }
}
